// VectorQuantizer_86225763435035
// MI455X (gfx1250) — matrix operand order repaired and checked by reading, not compiled, not run on hardware
//
#include <hip/hip_runtime.h>

typedef __attribute__((ext_vector_type(16))) _Float16 v16h;
typedef __attribute__((ext_vector_type(8)))  float    v8f;
typedef __attribute__((ext_vector_type(4)))  int      v4i;

#define B_  16
#define D_  128
#define T_  4096
#define K_  1024
#define M_TOTAL (B_ * T_)          // 65536 query rows
#define NT  (K_ / 16)              // 64 code tiles (N dimension)
#define KC  (D_ / 32)              // 4 k-chunks of 32 along D
#define ROWS_PER_WAVE 32           // two 16-row WMMA tiles per wave
#define NWAVES (M_TOTAL / ROWS_PER_WAVE)   // 2048

// workspace layout (bytes)
#define WHI_OFF  0
#define WLO_OFF  (WHI_OFF + (NT * KC * 32 * 8) * 4)   // 262144
#define WN_OFF   (WLO_OFF + (NT * KC * 32 * 8) * 4)   // 524288
#define PART_OFF (WN_OFF + K_ * 4)                    // 528384

// --------------------------- CDNA5 async helpers ---------------------------
__device__ __forceinline__ void async_b128(const void* g, void* l) {
#if __has_builtin(__builtin_amdgcn_global_load_async_to_lds_b128)
  __builtin_amdgcn_global_load_async_to_lds_b128((v4i*)g, (v4i*)l, 0, 0);
#else
  unsigned loff = (unsigned)(unsigned long long)l;   // low 32b of flat = LDS offset
  asm volatile("global_load_async_to_lds_b128 %0, %1, off"
               :: "v"(loff), "v"(g) : "memory");
#endif
}

__device__ __forceinline__ void wait_async0() {
#if __has_builtin(__builtin_amdgcn_s_wait_asynccnt)
  __builtin_amdgcn_s_wait_asynccnt(0);
#else
  asm volatile("s_wait_asynccnt 0x0" ::: "memory");
#endif
}

#define WMMA(A, Bm, C) \
  __builtin_amdgcn_wmma_f32_16x16x32_f16(false, (A), false, (Bm), (short)0, (C), false, false)

// ---------------------------------------------------------------------------
// Pre-swizzle W into the CDNA5 WMMA B-matrix (32x16 f16) layout, split hi/lo.
// Per-tile chunk is 4 KB contiguous: off = ((tile*4+kc)*32+lane)*16 halves.
// ---------------------------------------------------------------------------
__global__ __launch_bounds__(256) void vq_prep_w(const float* __restrict__ W,
                                                 _Float16* __restrict__ Whi,
                                                 _Float16* __restrict__ Wlo) {
  int tid  = blockIdx.x * 256 + threadIdx.x;      // 65536 total
  int v    = tid & 7;
  int lane = (tid >> 3) & 31;
  int kc   = (tid >> 8) & 3;
  int tile = tid >> 10;
  int n  = tile * 16 + (lane & 15);
  int k0 = ((lane < 16) ? 0 : 16) + 2 * v;
  int d  = kc * 32 + k0;
  float w0 = W[n * D_ + d];
  float w1 = W[n * D_ + d + 1];
  _Float16 h0 = (_Float16)w0, h1 = (_Float16)w1;
  Whi[tid * 2]     = h0;
  Whi[tid * 2 + 1] = h1;
  Wlo[tid * 2]     = (_Float16)(w0 - (float)h0);
  Wlo[tid * 2 + 1] = (_Float16)(w1 - (float)h1);
}

__global__ __launch_bounds__(256) void vq_wnorm(const float* __restrict__ W,
                                                float* __restrict__ wnorm) {
  int k = blockIdx.x * 256 + threadIdx.x;         // 1024 total
  float s = 0.f;
  for (int d = 0; d < D_; ++d) { float w = W[k * D_ + d]; s += w * w; }
  wnorm[k] = s;
}

// ---------------------------------------------------------------------------
// Main: one wave32 owns 32 rows (two 16x16 WMMA row tiles). B tiles staged in
// LDS once per workgroup with async loads (double buffered), shared by all 8
// waves. f16 hi/lo split (3 cross terms) -> ~fp32-accurate argmin.
// ---------------------------------------------------------------------------
__global__ __launch_bounds__(256) void vq_main(const float* __restrict__ x,
                                               const float* __restrict__ W,
                                               const _Float16* __restrict__ Whi,
                                               const _Float16* __restrict__ Wlo,
                                               const float* __restrict__ wnorm,
                                               float* __restrict__ qout,
                                               float* __restrict__ idxout,
                                               float* __restrict__ partials) {
  __shared__ __attribute__((aligned(32))) _Float16 sB[2][2][2048]; // [buf][hi/lo][4KB]
  __shared__ int sIdx[8 * ROWS_PER_WAVE];

  const int tid  = threadIdx.x;
  const int lane = tid & 31;
  const int wv   = tid >> 5;
  const int gw   = blockIdx.x * 8 + wv;     // wave id [0, 2048)
  const int m0   = gw * ROWS_PER_WAVE;
  const int b    = m0 >> 12;                // / T_
  const int t0   = m0 & (T_ - 1);
  const float* xb = x + (size_t)b * D_ * T_;

  const int tl    = lane & 15;              // row M within a 16-row tile
  const int khalf = (lane < 16) ? 0 : 8;

  // --- load both A tiles (2 x 16 x 128) in 16-bit A layout, split hi/lo ---
  v16h ahi0[KC], alo0[KC], ahi1[KC], alo1[KC];
#pragma unroll
  for (int kc = 0; kc < KC; ++kc) {
#pragma unroll
    for (int v = 0; v < 8; ++v) {
      int k = 2 * v + 2 * khalf;
      size_t base = (size_t)(kc * 32 + k) * T_ + t0 + tl;
      float x0 = xb[base];
      float x1 = xb[base + T_];
      float y0 = xb[base + 16];
      float y1 = xb[base + T_ + 16];
      _Float16 hx0 = (_Float16)x0, hx1 = (_Float16)x1;
      _Float16 hy0 = (_Float16)y0, hy1 = (_Float16)y1;
      ahi0[kc][2 * v] = hx0;  ahi0[kc][2 * v + 1] = hx1;
      ahi1[kc][2 * v] = hy0;  ahi1[kc][2 * v + 1] = hy1;
      alo0[kc][2 * v] = (_Float16)(x0 - (float)hx0);
      alo0[kc][2 * v + 1] = (_Float16)(x1 - (float)hx1);
      alo1[kc][2 * v] = (_Float16)(y0 - (float)hy0);
      alo1[kc][2 * v + 1] = (_Float16)(y1 - (float)hy1);
    }
  }

  float best0[8], best1[8];
  int   bidx0[8], bidx1[8];
#pragma unroll
  for (int v = 0; v < 8; ++v) {
    best0[v] = 3.4e38f; bidx0[v] = 0;
    best1[v] = 3.4e38f; bidx1[v] = 0;
  }

  // --- prologue: stage tile 0 into buffer 0 -------------------------------
  async_b128((const char*)Whi + tid * 16, (char*)&sB[0][0][0] + tid * 16);
  async_b128((const char*)Wlo + tid * 16, (char*)&sB[0][1][0] + tid * 16);
  wait_async0();
  __syncthreads();

  // --- sweep the 1024 codes in 16-wide tiles ------------------------------
  for (int tile = 0; tile < NT; ++tile) {
    int cur = tile & 1;
    if (tile + 1 < NT) {                    // async prefetch of next tile
      int nxt = cur ^ 1;
      size_t goff = (size_t)(tile + 1) * 4096 + tid * 16;
      async_b128((const char*)Whi + goff, (char*)&sB[nxt][0][0] + tid * 16);
      async_b128((const char*)Wlo + goff, (char*)&sB[nxt][1][0] + tid * 16);
    }

    v8f acc0 = {0.f, 0.f, 0.f, 0.f, 0.f, 0.f, 0.f, 0.f};
    v8f acc1 = {0.f, 0.f, 0.f, 0.f, 0.f, 0.f, 0.f, 0.f};
    const _Float16* bhB = &sB[cur][0][0];
    const _Float16* blB = &sB[cur][1][0];
#pragma unroll
    for (int kc = 0; kc < KC; ++kc) {
      v16h bh = *(const v16h*)(bhB + (kc * 32 + lane) * 16);
      v16h bl = *(const v16h*)(blB + (kc * 32 + lane) * 16);
      acc0 = WMMA(ahi0[kc], bh, acc0);
      acc0 = WMMA(ahi0[kc], bl, acc0);
      acc0 = WMMA(alo0[kc], bh, acc0);
      acc1 = WMMA(ahi1[kc], bh, acc1);
      acc1 = WMMA(ahi1[kc], bl, acc1);
      acc1 = WMMA(alo1[kc], bh, acc1);
    }

    int   n  = tile * 16 + tl;              // this lane's code column
    float wn = wnorm[n];
#pragma unroll
    for (int v = 0; v < 8; ++v) {
      float s0 = wn - 2.0f * acc0[v];       // ||w||^2 - 2 x.w
      float s1 = wn - 2.0f * acc1[v];
      if (s0 < best0[v]) { best0[v] = s0; bidx0[v] = n; }
      if (s1 < best1[v]) { best1[v] = s1; bidx1[v] = n; }
    }

    wait_async0();                          // my prefetch for tile+1 landed
    __syncthreads();                        // whole block done with cur & nxt ready
  }

  // --- argmin across the 16 lanes sharing each row (tie -> lowest index) --
#pragma unroll
  for (int mask = 8; mask >= 1; mask >>= 1) {
#pragma unroll
    for (int v = 0; v < 8; ++v) {
      float ob0 = __shfl_xor(best0[v], mask, 32);
      int   oi0 = __shfl_xor(bidx0[v], mask, 32);
      if (ob0 < best0[v] || (ob0 == best0[v] && oi0 < bidx0[v])) {
        best0[v] = ob0; bidx0[v] = oi0;
      }
      float ob1 = __shfl_xor(best1[v], mask, 32);
      int   oi1 = __shfl_xor(bidx1[v], mask, 32);
      if (ob1 < best1[v] || (ob1 == best1[v] && oi1 < bidx1[v])) {
        best1[v] = ob1; bidx1[v] = oi1;
      }
    }
  }

  if (lane == 0 || lane == 16) {
    int rbase = (lane == 0) ? 0 : 8;        // C/D layout: VGPR v -> M=v(+8)
#pragma unroll
    for (int v = 0; v < 8; ++v) {
      sIdx[wv * ROWS_PER_WAVE + rbase + v] = bidx0[v];
      sIdx[wv * ROWS_PER_WAVE + 16 + rbase + v] = bidx1[v];
      idxout[(size_t)b * T_ + t0 + rbase + v] = (float)bidx0[v];
      idxout[(size_t)b * T_ + t0 + 16 + rbase + v] = (float)bidx1[v];
    }
  }
  __syncthreads();

  // --- gather quantized output + commitment-loss partial ------------------
  float  lsum = 0.f;
  float* qb = qout + (size_t)b * D_ * T_;
#pragma unroll 4
  for (int i = 0; i < 128; ++i) {           // 32 t x 128 d / 32 lanes
    int flat = i * 32 + lane;
    int ttl  = flat & 31;
    int d    = flat >> 5;
    int idx  = sIdx[wv * ROWS_PER_WAVE + ttl];
    float wval = W[idx * D_ + d];
    float xv   = xb[(size_t)d * T_ + t0 + ttl];
    qb[(size_t)d * T_ + t0 + ttl] = wval;
    float df = wval - xv;
    lsum += df * df;
  }
#pragma unroll
  for (int mask = 16; mask >= 1; mask >>= 1) lsum += __shfl_xor(lsum, mask, 32);
  if (lane == 0) partials[gw] = lsum;
}

// deterministic fixed-order final reduction (no float atomics)
__global__ __launch_bounds__(256) void vq_loss(const float* __restrict__ partials,
                                               float* __restrict__ loss_out) {
  __shared__ float s[256];
  int t = threadIdx.x;
  float v = 0.f;
  for (int i = 0; i < NWAVES / 256; ++i) v += partials[t * (NWAVES / 256) + i];
  s[t] = v;
  __syncthreads();
  for (int stride = 128; stride > 0; stride >>= 1) {
    if (t < stride) s[t] += s[t + stride];
    __syncthreads();
  }
  if (t == 0) loss_out[0] = s[0] * (0.25f / (float)((size_t)B_ * D_ * T_));
}

extern "C" void kernel_launch(void* const* d_in, const int* in_sizes, int n_in,
                              void* d_out, int out_size, void* d_ws, size_t ws_size,
                              hipStream_t stream) {
  const float* x = (const float*)d_in[0];
  const float* W = (const float*)d_in[1];

  float* out   = (float*)d_out;
  float* qo    = out;                                   // (B, D, T)
  float* idxo  = out + (size_t)B_ * D_ * T_;            // (B, T) as float
  float* losso = idxo + (size_t)B_ * T_;                // scalar

  char* ws = (char*)d_ws;
  _Float16* Whi = (_Float16*)(ws + WHI_OFF);
  _Float16* Wlo = (_Float16*)(ws + WLO_OFF);
  float*    wn  = (float*)(ws + WN_OFF);
  float*    pt  = (float*)(ws + PART_OFF);

  vq_prep_w<<<256, 256, 0, stream>>>(W, Whi, Wlo);
  vq_wnorm<<<K_ / 256, 256, 0, stream>>>(W, wn);
  vq_main<<<NWAVES / 8, 256, 0, stream>>>(x, W, Whi, Wlo, wn, qo, idxo, pt);
  vq_loss<<<1, 256, 0, stream>>>(pt, losso);
}
